// Attention_20040317403473
// MI455X (gfx1250) — compile-verified, matrix operand order checked by reading, not run on hardware
//
#include <hip/hip_runtime.h>

typedef __bf16 bf16;
typedef __attribute__((ext_vector_type(8)))  __bf16 bf16x8;
typedef __attribute__((ext_vector_type(16))) __bf16 bf16x16;
typedef __attribute__((ext_vector_type(8)))  float  f32x8;

#define S_LEN 256
#define BATCH 128
#define DDIM  4096   // D (output features)
#define FDIM  8192   // 2D (input features of the Linear)

// ---------------------------------------------------------------------------
// Kernel 1: fused attention with online softmax (single pass over inp).
// One block per batch b, 256 threads; each thread owns 16 d-slots
// (d = (e/4)*1024 + 4*t + e%4, float4-coalesced). Writes concat = [weighted |
// hidden] as bf16 hi/lo split pairs for the WMMA GEMM.
// ---------------------------------------------------------------------------
__global__ __launch_bounds__(256) void attn_softmax_weighted(
    const float* __restrict__ inp,
    const float* __restrict__ hidden,
    bf16* __restrict__ concat_hi,
    bf16* __restrict__ concat_lo)
{
  const int b    = blockIdx.x;
  const int t    = threadIdx.x;
  const int wid  = t >> 5;
  const int lane = t & 31;

  __shared__ float sred[2][8];   // double-buffered cross-wave partials

  float h[16], acc[16];
  const float* hptr = hidden + (size_t)b * DDIM + 4 * t;
#pragma unroll
  for (int j = 0; j < 4; ++j) {
    float4 v = *(const float4*)(hptr + j * 1024);
    h[4 * j + 0] = v.x; h[4 * j + 1] = v.y;
    h[4 * j + 2] = v.z; h[4 * j + 3] = v.w;
  }
#pragma unroll
  for (int e = 0; e < 16; ++e) acc[e] = 0.f;

  float m = -3.402823466e38f;
  float l = 0.f;

  const float* base = inp + (size_t)b * DDIM + 4 * t;
  const size_t sstride = (size_t)BATCH * DDIM;

  for (int s = 0; s < S_LEN; ++s) {
    float x[16];
    const float* p = base + (size_t)s * sstride;
#pragma unroll
    for (int j = 0; j < 4; ++j) {
      float4 v = *(const float4*)(p + j * 1024);
      x[4 * j + 0] = v.x; x[4 * j + 1] = v.y;
      x[4 * j + 2] = v.z; x[4 * j + 3] = v.w;
    }

    // dot(inp[s,b,:], hidden[b,:]) : per-thread partial, wave32 + LDS reduce
    float part = 0.f;
#pragma unroll
    for (int e = 0; e < 16; ++e) part += x[e] * h[e];
#pragma unroll
    for (int off = 16; off > 0; off >>= 1)
      part += __shfl_xor(part, off, 32);

    if (lane == 0) sred[s & 1][wid] = part;
    __syncthreads();
    float score = 0.f;
#pragma unroll
    for (int w2 = 0; w2 < 8; ++w2) score += sred[s & 1][w2];

    // online softmax update
    const float mnew = fmaxf(m, score);
    const float corr = __expf(m - mnew);
    const float pr   = __expf(score - mnew);
    l = l * corr + pr;
#pragma unroll
    for (int e = 0; e < 16; ++e) acc[e] = acc[e] * corr + pr * x[e];
    m = mnew;
  }

  const float inv = 1.f / l;
  bf16* chi = concat_hi + (size_t)b * FDIM;
  bf16* clo = concat_lo + (size_t)b * FDIM;
#pragma unroll
  for (int e = 0; e < 16; ++e) {
    const int d = (e >> 2) * 1024 + 4 * t + (e & 3);
    const float wv = acc[e] * inv;          // weighted -> concat[:, 0:4096]
    const bf16 whi = (bf16)wv;
    chi[d] = whi;
    clo[d] = (bf16)(wv - (float)whi);
    const float hv = h[e];                  // hidden  -> concat[:, 4096:8192]
    const bf16 hhi = (bf16)hv;
    chi[DDIM + d] = hhi;
    clo[DDIM + d] = (bf16)(hv - (float)hhi);
  }
}

// ---------------------------------------------------------------------------
// Kernel 2: out = concat @ W^T + bias via bf16 WMMA with hi/lo split
// (3 x v_wmma_f32_16x16x32_bf16 per tile-K-step ~ fp32-quality accumulate).
// Grid: 64 blocks, each owns a 64-wide N strip (W read from HBM exactly once).
// Block: 256 threads = 8 waves; wave w computes M-tile rows [16w,16w+16) x
// 4 N-tiles. W f32 -> bf16 hi/lo staged in LDS per 32-wide K step.
// ---------------------------------------------------------------------------
__device__ inline bf16x16 make_frag(const bf16* p0, const bf16* p1) {
  bf16x8 a = *(const bf16x8*)p0;
  bf16x8 b = *(const bf16x8*)p1;
  bf16x16 r;
#pragma unroll
  for (int i = 0; i < 8; ++i) { r[i] = a[i]; r[i + 8] = b[i]; }
  return r;
}

__global__ __launch_bounds__(256) void ffn_gemm_wmma(
    const bf16*  __restrict__ Ahi,
    const bf16*  __restrict__ Alo,
    const float* __restrict__ W,
    const float* __restrict__ bias,
    float*       __restrict__ out)
{
  __shared__ __attribute__((aligned(16))) bf16 wh[64 * 32];
  __shared__ __attribute__((aligned(16))) bf16 wl[64 * 32];

  const int t     = threadIdx.x;
  const int wid   = t >> 5;
  const int lane  = t & 31;
  const int nbase = blockIdx.x * 64;
  const int mbase = wid * 16;
  const int lrow  = lane & 15;
  const int khalf = (lane >> 4) << 3;     // 0 or 8 (ISA 16-bit A/B lane split)

  f32x8 c[4];
#pragma unroll
  for (int q = 0; q < 4; ++q)
#pragma unroll
    for (int i = 0; i < 8; ++i) c[q][i] = 0.f;

  // W staging assignment: thread -> (row r in strip, 8-col chunk c0)
  const int r  = t >> 2;                  // 0..63
  const int c0 = (t & 3) << 3;            // 0,8,16,24
  const float* wrow = W + (size_t)(nbase + r) * FDIM + c0;
  const bf16*  ah_p = Ahi + (size_t)(mbase + lrow) * FDIM + khalf;
  const bf16*  al_p = Alo + (size_t)(mbase + lrow) * FDIM + khalf;
  bf16* whst = &wh[r * 32 + c0];
  bf16* wlst = &wl[r * 32 + c0];

  for (int k = 0; k < FDIM; k += 32) {
    // stage W[nstrip, k:k+32) as bf16 hi/lo into LDS
    float4 v0 = *(const float4*)(wrow + k);
    float4 v1 = *(const float4*)(wrow + k + 4);
    float f[8] = { v0.x, v0.y, v0.z, v0.w, v1.x, v1.y, v1.z, v1.w };
#pragma unroll
    for (int e = 0; e < 8; ++e) {
      const bf16 hi = (bf16)f[e];
      whst[e] = hi;
      wlst[e] = (bf16)(f[e] - (float)hi);
    }
    if (k + 32 < FDIM) __builtin_prefetch(wrow + k + 32, 0, 0);
    __syncthreads();

    // A fragments (global, bf16): lanes 0-15 take K 0..7 / 16..23,
    // lanes 16-31 take K 8..15 / 24..31 (ISA 16-bit A 16x32 layout)
    const bf16x16 ah = make_frag(ah_p + k, ah_p + k + 16);
    const bf16x16 al = make_frag(al_p + k, al_p + k + 16);

#pragma unroll
    for (int q = 0; q < 4; ++q) {
      const int brow = q * 16 + lrow;     // B column n = W row
      const bf16x16 bh = make_frag(&wh[brow * 32 + khalf],
                                   &wh[brow * 32 + khalf + 16]);
      const bf16x16 bl = make_frag(&wl[brow * 32 + khalf],
                                   &wl[brow * 32 + khalf + 16]);
      c[q] = __builtin_amdgcn_wmma_f32_16x16x32_bf16(false, ah, false, bh,
                                                     (short)0, c[q], false, false);
      c[q] = __builtin_amdgcn_wmma_f32_16x16x32_bf16(false, al, false, bh,
                                                     (short)0, c[q], false, false);
      c[q] = __builtin_amdgcn_wmma_f32_16x16x32_bf16(false, ah, false, bl,
                                                     (short)0, c[q], false, false);
    }
    __syncthreads();
  }

  // epilogue: C vgpr i -> M = i + 8*(lane>=16), N = lane&15
  const int row0 = mbase + ((lane >> 4) << 3);
#pragma unroll
  for (int q = 0; q < 4; ++q) {
    const int n = nbase + q * 16 + lrow;
    const float bv = bias[n];
#pragma unroll
    for (int i = 0; i < 8; ++i)
      out[(size_t)(row0 + i) * DDIM + n] = c[q][i] + bv;
  }
}

// ---------------------------------------------------------------------------
extern "C" void kernel_launch(void* const* d_in, const int* in_sizes, int n_in,
                              void* d_out, int out_size, void* d_ws, size_t ws_size,
                              hipStream_t stream) {
  (void)in_sizes; (void)n_in; (void)out_size; (void)ws_size;
  const float* inp    = (const float*)d_in[0];   // [256,128,4096]
  const float* hidden = (const float*)d_in[1];   // [1,128,4096]
  const float* W      = (const float*)d_in[2];   // [4096,8192]
  const float* bias   = (const float*)d_in[3];   // [4096]
  float* out = (float*)d_out;                    // [1,128,4096]

  // workspace: concat bf16 hi/lo, 128x8192 each (2 MB + 2 MB)
  bf16* chi = (bf16*)d_ws;
  bf16* clo = chi + (size_t)BATCH * FDIM;

  attn_softmax_weighted<<<dim3(BATCH), dim3(256), 0, stream>>>(inp, hidden, chi, clo);
  ffn_gemm_wmma<<<dim3(DDIM / 64), dim3(256), 0, stream>>>(chi, clo, W, bias, out);
}
